// WavLMEncoder_50525995270850
// MI455X (gfx1250) — matrix operand order repaired and checked by reading, not compiled, not run on hardware
//
#include <hip/hip_runtime.h>
#include <hip/hip_bf16.h>
#include <math.h>

typedef __bf16 bf16_t;
typedef __attribute__((ext_vector_type(8)))  bf16_t bf16x8;
typedef __attribute__((ext_vector_type(16))) bf16_t bf16x16;
typedef __attribute__((ext_vector_type(8)))  float  f32x8;

#define DEV __device__ __forceinline__

DEV float lrelu_f(float v){ return v > 0.f ? v : 0.1f*v; }
DEV float act_apply(float v, int act){
  if (act == 1) return lrelu_f(v);
  if (act == 2) return 0.5f*v*(1.f + erff(v*0.70710678118654752f));   // exact GELU
  if (act == 3) return v/(1.f + __expf(-v));                          // SiLU
  return v;
}

// ---------------------------------------------------------------------------
// WMMA GEMM:  C[M,N] = A[M,K32] * W[N,K32]^T + bias, fused activation.
// A, W bf16 row-major with K padded to multiple of 32 (zero fill).
// One wave computes a 32x32 tile with 4 f32 accumulators via
// v_wmma_f32_16x16x32_bf16.
// ---------------------------------------------------------------------------
DEV bf16x16 load_a_frag(const bf16_t* p){
  // A 16x32 bf16 layout: lane (r,h): elems 0..7 -> K=h*8+{0..7}, elems 8..15 -> K=16+h*8+{0..7}
  bf16x8 lo = *(const bf16x8*)p;
  bf16x8 hi = *(const bf16x8*)(p + 16);
  return __builtin_shufflevector(lo, hi, 0,1,2,3,4,5,6,7,8,9,10,11,12,13,14,15);
}

__global__ __launch_bounds__(256) void gemm_wmma_bf16(
    const bf16_t* __restrict__ A, const bf16_t* __restrict__ W,
    const float* __restrict__ bias, float* __restrict__ C,
    int Mpad, int Mvalid, int N, int K32, int act){
  int wid = (int)((blockIdx.x*blockDim.x + threadIdx.x) >> 5);
  int Nt = N >> 5;
  int Mt = Mpad >> 5;
  if (wid >= Mt*Nt) return;
  int tm = wid / Nt, tn = wid % Nt;
  int lane = threadIdx.x & 31;
  int r = lane & 15, h = lane >> 4;

  const bf16_t* Ab = A + ((size_t)(tm*32 + r))*K32 + h*8;   // A fragment base
  const bf16_t* Wb = W + ((size_t)(tn*32 + r))*K32 + h*8;   // B fragment base (lane r = column)
  size_t rowStep = (size_t)16*K32;

  f32x8 c00 = {0.f,0.f,0.f,0.f,0.f,0.f,0.f,0.f};
  f32x8 c01 = c00, c10 = c00, c11 = c00;

  for (int k0 = 0; k0 < K32; k0 += 32){
    bf16x16 a0 = load_a_frag(Ab + k0);
    bf16x16 a1 = load_a_frag(Ab + rowStep + k0);
    bf16x16 b0 = load_a_frag(Wb + k0);
    bf16x16 b1 = load_a_frag(Wb + rowStep + k0);
    __builtin_prefetch(Ab + k0 + 128, 0, 1);   // global_prefetch_b8 (K block 4 iterations ahead)
    __builtin_prefetch(Wb + k0 + 128, 0, 1);
    c00 = __builtin_amdgcn_wmma_f32_16x16x32_bf16(false, a0, false, b0, (short)0, c00, false, false);
    c01 = __builtin_amdgcn_wmma_f32_16x16x32_bf16(false, a0, false, b1, (short)0, c01, false, false);
    c10 = __builtin_amdgcn_wmma_f32_16x16x32_bf16(false, a1, false, b0, (short)0, c10, false, false);
    c11 = __builtin_amdgcn_wmma_f32_16x16x32_bf16(false, a1, false, b1, (short)0, c11, false, false);
  }

  // C/D layout: vgpr vr -> M = vr + 8*h ; N = lane&15
  int n0 = tn*32 + r;
  float bi0 = bias ? bias[n0]      : 0.f;
  float bi1 = bias ? bias[n0 + 16] : 0.f;
  #pragma unroll
  for (int vr = 0; vr < 8; ++vr){
    int m0 = tm*32 + vr + h*8;
    int m1 = m0 + 16;
    if (m0 < Mvalid){
      C[(size_t)m0*N + n0]      = act_apply(c00[vr] + bi0, act);
      C[(size_t)m0*N + n0 + 16] = act_apply(c01[vr] + bi1, act);
    }
    if (m1 < Mvalid){
      C[(size_t)m1*N + n0]      = act_apply(c10[vr] + bi0, act);
      C[(size_t)m1*N + n0 + 16] = act_apply(c11[vr] + bi1, act);
    }
  }
}

// ---------------------------------------------------------------------------
// im2col: channels-last f32 activations [B*Tin, Cin] -> bf16 [Mpad, K32]
// column order = (kw, ci). Fused pre-activation (act==1 : leaky 0.1).
// Also used with Kw=1 as plain f32->bf16 conversion for linear layers.
// ---------------------------------------------------------------------------
__global__ void im2col_k(const float* __restrict__ src, bf16_t* __restrict__ dst,
                         int Bn, int Tin, int Cin, int Tout, int Kw, int stride,
                         int pad, int dil, int K32, int Mpad, int act){
  size_t idx = (size_t)blockIdx.x*blockDim.x + threadIdx.x;
  size_t total = (size_t)Mpad*K32;
  if (idx >= total) return;
  int col = (int)(idx % K32);
  int m   = (int)(idx / K32);
  float v = 0.f;
  int K = Kw*Cin;
  if (m < Bn*Tout && col < K){
    int b = m / Tout, t = m % Tout;
    int kw = col / Cin, ci = col % Cin;
    int tin = t*stride - pad + kw*dil;
    if (tin >= 0 && tin < Tin) v = src[((size_t)b*Tin + tin)*Cin + ci];
    if (act == 1) v = lrelu_f(v);
  }
  dst[idx] = (bf16_t)v;
}

// Weight convert: f32 [O, I, Kw] (torch OIH; Kw=1 for linear [O,I]) -> bf16 [O, K32], col=(kw,i)
__global__ void wconv_k(const float* __restrict__ w, bf16_t* __restrict__ dst,
                        int O, int I, int Kw, int K32){
  size_t idx = (size_t)blockIdx.x*blockDim.x + threadIdx.x;
  size_t total = (size_t)O*K32;
  if (idx >= total) return;
  int col = (int)(idx % K32);
  int o   = (int)(idx / K32);
  float v = 0.f;
  if (col < I*Kw){
    int kw = col / I, i = col % I;
    v = w[((size_t)o*I + i)*Kw + kw];
  }
  dst[idx] = (bf16_t)v;
}

// ---------------------------------------------------------------------------
// Elementwise / small kernels
// ---------------------------------------------------------------------------
__global__ void add_k(float* __restrict__ x, const float* __restrict__ y, float s, size_t n){
  size_t i = (size_t)blockIdx.x*blockDim.x + threadIdx.x;
  if (i < n) x[i] += s*y[i];
}
__global__ void copy_k(const float* __restrict__ src, float* __restrict__ dst, size_t n){
  size_t i = (size_t)blockIdx.x*blockDim.x + threadIdx.x;
  if (i < n) dst[i] = src[i];
}
__global__ void snake_k(float* __restrict__ x, const float* __restrict__ ap, size_t n, int C){
  size_t i = (size_t)blockIdx.x*blockDim.x + threadIdx.x;
  if (i >= n) return;
  int c = (int)(i % C);
  float v = ap[c];
  float sp = v > 20.f ? v : log1pf(__expf(v));
  float a = sp + 0.01f;
  float xv = x[i];
  float sn = __sinf(a*xv);
  x[i] = xv + sn*sn / fmaxf(a, 0.01f);
}
__global__ void mask_k(float* __restrict__ x, const int* __restrict__ mask,
                       const float* __restrict__ emb, int BT, int C){
  size_t i = (size_t)blockIdx.x*blockDim.x + threadIdx.x;
  if (i >= (size_t)BT*C) return;
  int row = (int)(i / C), c = (int)(i % C);
  float m = (float)mask[row];
  x[i] = x[i]*m + emb[c]*(1.f - m);
}
__global__ void glu_k(const float* __restrict__ in, float* __restrict__ out, int BT, int C){
  size_t i = (size_t)blockIdx.x*blockDim.x + threadIdx.x;
  if (i >= (size_t)BT*C) return;
  int row = (int)(i / C), c = (int)(i % C);
  float a = in[(size_t)row*2*C + c];
  float g = in[(size_t)row*2*C + C + c];
  out[i] = a / (1.f + __expf(-g)) * 1.f;  // a * sigmoid(g)
}
__global__ void dwconv_k(const float* __restrict__ x, const float* __restrict__ w,
                         const float* __restrict__ bias, float* __restrict__ y,
                         int Bn, int T, int C, int Kw, int pad){
  size_t idx = (size_t)blockIdx.x*blockDim.x + threadIdx.x;
  if (idx >= (size_t)Bn*T*C) return;
  int c = (int)(idx % C);
  int t = (int)((idx / C) % T);
  int b = (int)(idx / ((size_t)T*C));
  float s = bias[c];
  for (int k = 0; k < Kw; ++k){
    int tt = t + k - pad;
    if (tt >= 0 && tt < T) s += w[c*Kw + k] * x[((size_t)(b*T + tt))*C + c];
  }
  y[idx] = s;
}
// GroupNorm(1,C): reduce per batch over T*C contiguous floats
__global__ void gn_reduce_k(const float* __restrict__ x, float* __restrict__ stats, int TC){
  __shared__ float sh1[256], sh2[256];
  int b = blockIdx.x;
  const float* p = x + (size_t)b*TC;
  float s = 0.f, s2 = 0.f;
  for (int i = threadIdx.x; i < TC; i += blockDim.x){ float v = p[i]; s += v; s2 += v*v; }
  sh1[threadIdx.x] = s; sh2[threadIdx.x] = s2; __syncthreads();
  for (int o = 128; o; o >>= 1){
    if ((int)threadIdx.x < o){ sh1[threadIdx.x] += sh1[threadIdx.x+o]; sh2[threadIdx.x] += sh2[threadIdx.x+o]; }
    __syncthreads();
  }
  if (threadIdx.x == 0){
    float mean = sh1[0]/TC;
    float var  = sh2[0]/TC - mean*mean;
    stats[b*2] = mean; stats[b*2+1] = rsqrtf(var + 1e-5f);
  }
}
__global__ void gn_apply_k(const float* __restrict__ x, float* __restrict__ y,
                           const float* __restrict__ g, const float* __restrict__ be,
                           const float* __restrict__ stats, int T, int C, int act){
  size_t i = (size_t)blockIdx.x*blockDim.x + threadIdx.x;
  if (i >= (size_t)2*T*C) return;
  int row = (int)(i / C), c = (int)(i % C);
  int b = row / T;
  float v = (x[i] - stats[b*2]) * stats[b*2+1] * g[c] + be[c];
  y[i] = act_apply(v, act);
}
// Attention gates: g = sigmoid(q . gate[h]) per (b,t,h); stored [B,H,T]
__global__ void gates_k(const float* __restrict__ qkv, const float* __restrict__ wu,
                        const float* __restrict__ ww, float* __restrict__ gu,
                        float* __restrict__ gw, int Bn, int T, int H){
  int idx = (int)(blockIdx.x*blockDim.x + threadIdx.x);
  if (idx >= Bn*T*H) return;
  int h = idx % H, t = (idx / H) % T, b = idx / (H*T);
  const float* qp = qkv + ((size_t)(b*T + t))*1536 + h*16;
  float su = 0.f, sw = 0.f;
  #pragma unroll
  for (int d = 0; d < 16; ++d){ float q = qp[d]; su += q*wu[h*16+d]; sw += q*ww[h*16+d]; }
  size_t o = ((size_t)(b*H) + h)*T + t;
  gu[o] = 1.f/(1.f + __expf(-su));
  gw[o] = 1.f/(1.f + __expf(-sw));
}
DEV int bucket_fn(int i, int j){
  int rel = j - i; int sg = rel >= 0 ? 1 : 0; int ab = rel >= 0 ? rel : -rel;
  int bkt;
  if (ab < 80) bkt = ab;
  else {
    float lp = 80.0f + __logf((float)ab * (1.0f/80.0f)) * (80.0f/__logf(10.0f));
    bkt = (int)lp; if (bkt > 159) bkt = 159;
  }
  bkt += sg*160;
  if (bkt < 0) bkt = 0; if (bkt > 319) bkt = 319;
  return bkt;
}
// One wave per (b,h,query); stable softmax of qk/4 + bias with LDS-staged scores.
__global__ __launch_bounds__(256) void attn_k(const float* __restrict__ qkv,
    const float* __restrict__ emb, const float* __restrict__ scale,
    const float* __restrict__ gu, const float* __restrict__ gw,
    float* __restrict__ out, int Bn, int T, int H){
  __shared__ float ssc[8][512];
  int wv = threadIdx.x >> 5, lane = threadIdx.x & 31;
  int id = blockIdx.x*8 + wv;
  if (id >= Bn*H*T) return;
  int i = id % T, h = (id / T) % H, b = id / (T*H);
  const float* qp = qkv + ((size_t)(b*T + i))*1536 + h*16;
  float q[16];
  #pragma unroll
  for (int d = 0; d < 16; ++d) q[d] = qp[d];
  float gui = gu[((size_t)(b*H) + h)*T + i];
  float gwi = gw[((size_t)(b*H) + h)*T + i];
  float fi = 1.f + gui + (1.f - gui)*scale[h]*gwi;
  float mloc = -3.4e38f;
  for (int j = lane; j < T; j += 32){
    const float* kp = qkv + ((size_t)(b*T + j))*1536 + 512 + h*16;
    float s = 0.f;
    #pragma unroll
    for (int d = 0; d < 16; ++d) s += q[d]*kp[d];
    s *= 0.25f;                                   // 1/sqrt(hd)
    s += emb[bucket_fn(i, j)*H + h] * fi;         // rel-pos gated bias
    ssc[wv][j] = s;
    mloc = fmaxf(mloc, s);
  }
  for (int o = 16; o; o >>= 1) mloc = fmaxf(mloc, __shfl_xor(mloc, o, 32));
  float den = 0.f, acc[16];
  #pragma unroll
  for (int d = 0; d < 16; ++d) acc[d] = 0.f;
  for (int j = lane; j < T; j += 32){
    float p = __expf(ssc[wv][j] - mloc);
    den += p;
    const float* vp = qkv + ((size_t)(b*T + j))*1536 + 1024 + h*16;
    #pragma unroll
    for (int d = 0; d < 16; ++d) acc[d] += p*vp[d];
  }
  for (int o = 16; o; o >>= 1){
    den += __shfl_xor(den, o, 32);
    #pragma unroll
    for (int d = 0; d < 16; ++d) acc[d] += __shfl_xor(acc[d], o, 32);
  }
  if (lane == 0){
    float inv = 1.f/den;
    float* op = out + ((size_t)(b*T + i))*512 + h*16;
    #pragma unroll
    for (int d = 0; d < 16; ++d) op[d] = acc[d]*inv;
  }
}

// ---------------------------------------------------------------------------
// Host orchestration
// ---------------------------------------------------------------------------
static inline int ru32(int x){ return (x + 31) & ~31; }
static inline unsigned gblk(size_t n){ return (unsigned)((n + 255) / 256); }

extern "C" void kernel_launch(void* const* d_in, const int* in_sizes, int n_in,
                              void* d_out, int out_size, void* d_ws, size_t ws_size,
                              hipStream_t stream){
  (void)in_sizes; (void)n_in; (void)out_size; (void)ws_size;
  auto F = [&](int i){ return (const float*)d_in[i]; };

  // --- param leaf indices (wav, mask, then jax pytree sorted-key order) ---
  const int IDX_WAV = 0, IDX_MASK = 1;
  auto conf = [&](int l){ return 2 + l*22; };
  // conformer offsets: conv: dwb0 dww1 gn1b2 gn1g3 gn2b4 gn2g5 pw1b6 pw1w7 pw2b8 pw2w9
  //                    ff1: b1=10 b2=11 w1=12 w2=13 ; ff2: 14..17 ; out: b18 w19 ; qkv: b20 w21
  const int ENC0 = 2 + 4*22;                              // 90
  auto sbase = [&](int s){ return ENC0 + s*51; };
  auto ridx  = [&](int s,int r,int which,int d,int bw){ return sbase(s) + 2 + r*12 + which*6 + d*2 + bw; };
  const int IDX_FPB = 243, IDX_FPW = 244, IDX_ICB = 245, IDX_ICW = 246, IDX_MEMB = 247;
  const int IDX_PRB = 248, IDX_PRW = 249;
  const int IDX_EMB = 250, IDX_GUW = 251, IDX_GWW = 252, IDX_SCALE = 253;

  // --- workspace carve (ws >= ~176MB assumed) ---
  char* base = (char*)d_ws;
  auto carve = [&](size_t bytes){ char* p = base; base += (bytes + 255) & ~(size_t)255; return p; };
  float*  PA   = (float*)carve((size_t)42 << 20);
  float*  PB   = (float*)carve((size_t)42 << 20);
  float*  PC   = (float*)carve((size_t)42 << 20);
  bf16_t* COL  = (bf16_t*)carve((size_t)42 << 20);
  bf16_t* WT   = (bf16_t*)carve((size_t)4  << 20);
  float*  GU   = (float*)carve((size_t)2*32*499*4);
  float*  GW   = (float*)carve((size_t)2*32*499*4);
  float*  STATS= (float*)carve(256);

  float *cur = PA, *t1 = PB, *t2 = PC;
  auto swapp = [](float*& a, float*& b){ float* t = a; a = b; b = t; };

  // --- helpers ---
  auto run_gemm = [&](const bf16_t* A, int widx, int bidx, int O, int I, int Kw,
                      float* out, int Mreal, int act){
    int K32 = ru32(I*Kw);
    size_t wn = (size_t)O*K32;
    wconv_k<<<gblk(wn), 256, 0, stream>>>(F(widx), WT, O, I, Kw, K32);
    int Mpad = ru32(Mreal);
    size_t waves = (size_t)(Mpad/32) * (O/32);
    gemm_wmma_bf16<<<gblk(waves*32), 256, 0, stream>>>(A, WT, bidx >= 0 ? F(bidx) : nullptr,
                                                       out, Mpad, Mreal, O, K32, act);
  };
  auto do_im2col = [&](const float* src, int Tin, int Cin, int Tout, int Kw, int stride,
                       int pad, int dil, int act){
    int K32 = ru32(Cin*Kw), Mpad = ru32(2*Tout);
    size_t n = (size_t)Mpad*K32;
    im2col_k<<<gblk(n), 256, 0, stream>>>(src, COL, 2, Tin, Cin, Tout, Kw, stride, pad, dil,
                                          K32, Mpad, act);
  };
  auto convgemm = [&](const float* src, int Tin, int Cin, int Tout, int Kw, int stride,
                      int pad, int dil, int actpre, int widx, int bidx, int Cout,
                      float* out, int actpost){
    do_im2col(src, Tin, Cin, Tout, Kw, stride, pad, dil, actpre);
    run_gemm(COL, widx, bidx, Cout, Cin, Kw, out, 2*Tout, actpost);
  };
  auto tobf16 = [&](const float* src, int C){ do_im2col(src, 499, C, 499, 1, 1, 0, 1, 0); };
  auto resadd = [&](float* x, const float* y, float s, size_t n){
    add_k<<<gblk(n), 256, 0, stream>>>(x, y, s, n);
  };

  // ================= Encoder =================
  // input conv: [B,1,160000] -> [B*160000, 32], K=7 pad 3
  convgemm(F(IDX_WAV), 160000, 1, 160000, 7, 1, 3, 1, 0, IDX_ICW, IDX_ICB, 32, cur, 0);
  int T = 160000, C = 32;
  const int strides[3] = {8, 8, 5};
  const int chs[4] = {32, 64, 128, 256};
  const int p3[4] = {1, 3, 9, 27}, p5[4] = {1, 5, 25, 125};
  for (int s = 0; s < 3; ++s){
    int st = strides[s], cout = chs[s+1];
    int Tout = (T + 2*(st/2) - 2*st)/st + 1;
    convgemm(cur, T, C, Tout, 2*st, st, st/2, 1, 0, sbase(s)+1, sbase(s)+0, cout, t1, 0);
    snake_k<<<gblk((size_t)2*Tout*cout), 256, 0, stream>>>(t1, F(sbase(s)+50),
                                                           (size_t)2*Tout*cout, cout);
    swapp(cur, t1);
    T = Tout; C = cout;
    size_t nel = (size_t)2*T*C;
    for (int r = 0; r < 4; ++r){
      int dils[3] = {1, p3[r], p5[r]};
      for (int d = 0; d < 3; ++d){
        convgemm(cur, T, C, T, 3, 1, dils[d], dils[d], 1, ridx(s,r,0,d,1), ridx(s,r,0,d,0), C, t1, 0);
        convgemm(t1,  T, C, T, 3, 1, 1,       1,       1, ridx(s,r,1,d,1), ridx(s,r,1,d,0), C, t2, 0);
        resadd(cur, t2, 1.0f, nel);
      }
    }
  }
  // proj 1x1: 256 -> 512, then mask embedding
  convgemm(cur, 499, 256, 499, 1, 1, 0, 1, 0, IDX_PRW, IDX_PRB, 512, t1, 0);
  mask_k<<<gblk((size_t)998*512), 256, 0, stream>>>(t1, (const int*)d_in[IDX_MASK],
                                                    F(IDX_MEMB), 998, 512);
  swapp(cur, t1);

  // ================= Conformers =================
  const size_t ND = (size_t)998*512;
  for (int l = 0; l < 4; ++l){
    int cb = conf(l);
    // FF1 (half-step)
    tobf16(cur, 512);  run_gemm(COL, cb+12, cb+10, 2048, 512, 1, t1, 998, 2);
    tobf16(t1, 2048);  run_gemm(COL, cb+13, cb+11, 512, 2048, 1, t2, 998, 0);
    resadd(cur, t2, 0.5f, ND);
    // Self-attention with gated relative position bias
    tobf16(cur, 512);  run_gemm(COL, cb+21, cb+20, 1536, 512, 1, t1, 998, 0);
    gates_k<<<gblk((size_t)2*499*32), 256, 0, stream>>>(t1, F(IDX_GUW), F(IDX_GWW), GU, GW, 2, 499, 32);
    attn_k<<<gblk((size_t)2*32*499*32), 256, 0, stream>>>(t1, F(IDX_EMB), F(IDX_SCALE), GU, GW, t2, 2, 499, 32);
    tobf16(t2, 512);   run_gemm(COL, cb+19, cb+18, 512, 512, 1, t1, 998, 0);
    resadd(cur, t1, 1.0f, ND);
    // Convolution module
    gn_reduce_k<<<2, 256, 0, stream>>>(cur, STATS, 499*512);
    gn_apply_k<<<gblk(ND), 256, 0, stream>>>(cur, t2, F(cb+3), F(cb+2), STATS, 499, 512, 0);
    tobf16(t2, 512);   run_gemm(COL, cb+7, cb+6, 1024, 512, 1, t1, 998, 0);
    glu_k<<<gblk(ND), 256, 0, stream>>>(t1, t2, 998, 512);
    dwconv_k<<<gblk(ND), 256, 0, stream>>>(t2, F(cb+1), F(cb+0), t1, 2, 499, 512, 31, 15);
    gn_reduce_k<<<2, 256, 0, stream>>>(t1, STATS, 499*512);
    gn_apply_k<<<gblk(ND), 256, 0, stream>>>(t1, t2, F(cb+5), F(cb+4), STATS, 499, 512, 3);
    tobf16(t2, 512);   run_gemm(COL, cb+9, cb+8, 512, 512, 1, t1, 998, 0);
    resadd(cur, t1, 1.0f, ND);
    // FF2 (half-step)
    tobf16(cur, 512);  run_gemm(COL, cb+16, cb+14, 2048, 512, 1, t1, 998, 2);
    tobf16(t1, 2048);  run_gemm(COL, cb+17, cb+15, 512, 2048, 1, t2, 998, 0);
    resadd(cur, t2, 0.5f, ND);
  }

  // ================= Outputs: logits [B,T,1024] then hidden x [B,T,512] =================
  float* dout = (float*)d_out;
  tobf16(cur, 512);
  run_gemm(COL, IDX_FPW, IDX_FPB, 1024, 512, 1, dout, 998, 0);
  copy_k<<<gblk(ND), 256, 0, stream>>>(cur, dout + (size_t)998*1024, ND);
}
